// DecoderLayer_11622181503140
// MI455X (gfx1250) — matrix operand order repaired and checked by reading, not compiled, not run on hardware
//
#include <hip/hip_runtime.h>
#include <hip/hip_bf16.h>
#include <math.h>

typedef __attribute__((ext_vector_type(16))) _Float16 v16h;
typedef __attribute__((ext_vector_type(8)))  _Float16 v8h;
typedef __attribute__((ext_vector_type(8)))  float    v8f;
typedef __attribute__((ext_vector_type(2)))  float    f2;

#define N_NODES 8192
#define KNBR    48
#define CF      128
#define CEF     384
#define CIN     512     // CF + CEF
#define SCALE_F 30.0f
#define EPSF    1e-5f

__device__ __forceinline__ float gelu_exact(float x) {
    return 0.5f * x * (1.0f + erff(x * 0.70710678118654752f));
}

__device__ __forceinline__ void st_h2(_Float16* p, float a, float b) {
    union { _Float16 h[2]; unsigned u; } t;
    t.h[0] = (_Float16)a; t.h[1] = (_Float16)b;
    *(unsigned*)p = t.u;
}

union AFrag { v16h v; v8h h[2]; };

// ---------------------------------------------------------------------------
// Pack a weight matrix W (Ndim x Kdim, row-major; GEMM uses B[k][n] = W[n][k])
// into WMMA-B fragment order: fragment (kb, nb) stores, for lane l, half h:
//   n = nb*16 + (l & 15);  k = kb*32 + (h < 8 ? h : h + 8) + (l < 16 ? 0 : 8)
// so each lane's 16 halves (32 bytes) are contiguous -> one coalesced load.
// ---------------------------------------------------------------------------
__global__ __launch_bounds__(256) void pack_weights(const float* __restrict__ W,
                                                    _Float16* __restrict__ dst,
                                                    int Kdim, int Ndim) {
    int idx = blockIdx.x * 256 + threadIdx.x;
    int total = Kdim * Ndim;
    if (idx >= total) return;
    int frag = idx >> 9;       // 512 halves per fragment
    int r    = idx & 511;
    int lane = r >> 4;
    int h    = r & 15;
    int nbCount = Ndim >> 4;
    int kb = frag / nbCount;
    int nb = frag % nbCount;
    int n  = nb * 16 + (lane & 15);
    int k  = kb * 32 + ((h < 8) ? h : h + 8) + ((lane < 16) ? 0 : 8);
    dst[idx] = (_Float16)W[(size_t)n * Kdim + k];
}

// ---------------------------------------------------------------------------
// Kernel A: fused edge-message MLP (512->128->128->128, exact gelu),
// attention-masked neighbor reduction, residual + LayerNorm1.
// TWO nodes (96 edge rows) per workgroup (8 wave32) to halve L2 weight
// traffic. Output: h_v as f16 [N, 128].
// ---------------------------------------------------------------------------
__global__ __launch_bounds__(256) void msg_kernel(
    const float* __restrict__ nodef, const float* __restrict__ edgef,
    const float* __restrict__ attn,
    const _Float16* __restrict__ pW1, const float* __restrict__ b1,
    const _Float16* __restrict__ pW2, const float* __restrict__ b2,
    const _Float16* __restrict__ pW3, const float* __restrict__ b3,
    const float* __restrict__ ln1w, const float* __restrict__ ln1b,
    _Float16* __restrict__ hv)
{
    extern __shared__ char smem_raw[];
    const int ASTR = 520;   // 96 x 520 halves : concat(node, edge) f16
    const int MSTR = 136;   // 96 x 136 halves : intermediate messages
    _Float16* As   = (_Float16*)smem_raw;
    _Float16* M1   = As + 96 * ASTR;
    _Float16* M2   = M1 + 96 * MSTR;
    float*    red  = (float*)(M2 + 96 * MSTR);  // 256 pre-LN values (2 nodes)
    float*    aux  = red + 256;                 // 20 scratch floats
    float*    attw = aux + 20;                  // 96 attention weights

    const int n0   = blockIdx.x * 2;
    const int tid  = threadIdx.x;
    const int w    = tid >> 5;
    const int lane = tid & 31;
    const int colL = lane & 15;
    const int col  = w * 16 + colL;
    const int rsel = (lane < 16) ? 0 : 8;   // row offset in C/D frags, K offset in A frags

    // ---- stage concat(node_features, edge_features) -> LDS f16 -------------
    for (int p = tid; p < 96 * 256; p += 256) {
        int r  = p >> 8;                     // 0..95
        int c0 = (p & 255) * 2;
        int nd = (r >= KNBR) ? 1 : 0;
        int rr = r - nd * KNBR;
        f2 v;
        if (c0 < CF)
            v = *(const f2*)(nodef + (size_t)(n0 + nd) * CF + c0);
        else
            v = __builtin_nontemporal_load(
                    (const f2*)(edgef + ((size_t)(n0 + nd) * KNBR + rr) * CEF + (c0 - CF)));
        st_h2(&As[r * ASTR + c0], v[0], v[1]);
    }
    if (tid < 96) attw[tid] = attn[(size_t)n0 * KNBR + tid];   // contiguous 2-node rows
    __syncthreads();

    // ---- GEMM1: [96 x 512] @ [512 x 128], wave owns 16-col strip ----------
    {
        v8f acc[6] = {};
        #pragma unroll
        for (int kb = 0; kb < 16; ++kb) {
            v16h b = *(const v16h*)(pW1 + (((kb << 3) + w) << 9) + lane * 16);
            #pragma unroll
            for (int mt = 0; mt < 6; ++mt) {
                AFrag a;
                int row  = mt * 16 + colL;
                int off0 = row * ASTR + kb * 32 + rsel;
                a.h[0] = *(const v8h*)(As + off0);
                a.h[1] = *(const v8h*)(As + off0 + 16);
                acc[mt] = __builtin_amdgcn_wmma_f32_16x16x32_f16(
                    false, a.v, false, b, (short)0, acc[mt], false, false);
            }
        }
        float bias = b1[col];
        #pragma unroll
        for (int mt = 0; mt < 6; ++mt)
            #pragma unroll
            for (int i = 0; i < 8; ++i) {
                int row = mt * 16 + i + rsel;
                M1[row * MSTR + col] = (_Float16)gelu_exact(acc[mt][i] + bias);
            }
    }
    __syncthreads();

    // ---- GEMM2: [96 x 128] @ [128 x 128] ----------------------------------
    {
        v8f acc[6] = {};
        #pragma unroll
        for (int kb = 0; kb < 4; ++kb) {
            v16h b = *(const v16h*)(pW2 + (((kb << 3) + w) << 9) + lane * 16);
            #pragma unroll
            for (int mt = 0; mt < 6; ++mt) {
                AFrag a;
                int row  = mt * 16 + colL;
                int off0 = row * MSTR + kb * 32 + rsel;
                a.h[0] = *(const v8h*)(M1 + off0);
                a.h[1] = *(const v8h*)(M1 + off0 + 16);
                acc[mt] = __builtin_amdgcn_wmma_f32_16x16x32_f16(
                    false, a.v, false, b, (short)0, acc[mt], false, false);
            }
        }
        float bias = b2[col];
        #pragma unroll
        for (int mt = 0; mt < 6; ++mt)
            #pragma unroll
            for (int i = 0; i < 8; ++i) {
                int row = mt * 16 + i + rsel;
                M2[row * MSTR + col] = (_Float16)gelu_exact(acc[mt][i] + bias);
            }
    }
    __syncthreads();

    // ---- GEMM3 + attention mask + per-node neighbor reduction -------------
    {
        v8f acc[6] = {};
        #pragma unroll
        for (int kb = 0; kb < 4; ++kb) {
            v16h b = *(const v16h*)(pW3 + (((kb << 3) + w) << 9) + lane * 16);
            #pragma unroll
            for (int mt = 0; mt < 6; ++mt) {
                AFrag a;
                int row  = mt * 16 + colL;
                int off0 = row * MSTR + kb * 32 + rsel;
                a.h[0] = *(const v8h*)(M2 + off0);
                a.h[1] = *(const v8h*)(M2 + off0 + 16);
                acc[mt] = __builtin_amdgcn_wmma_f32_16x16x32_f16(
                    false, a.v, false, b, (short)0, acc[mt], false, false);
            }
        }
        float bias = b3[col];
        float p0 = 0.0f, p1 = 0.0f;
        #pragma unroll
        for (int mt = 0; mt < 6; ++mt)
            #pragma unroll
            for (int i = 0; i < 8; ++i) {
                int row = mt * 16 + i + rsel;
                float v = attw[row] * (acc[mt][i] + bias);
                if (mt < 3) p0 += v; else p1 += v;
            }
        p0 += __shfl_xor(p0, 16);   // fold lane halves (same col)
        p1 += __shfl_xor(p1, 16);
        if (lane < 16) {
            red[col]       = nodef[(size_t)n0 * CF + col]       + p0 * (1.0f / SCALE_F);
            red[CF + col]  = nodef[(size_t)(n0 + 1) * CF + col] + p1 * (1.0f / SCALE_F);
        }
    }
    __syncthreads();

    // ---- LayerNorm1: both nodes in parallel (waves 0-3 / 4-7) -------------
    float x = red[tid];
    float s = x, q = x * x;
    for (int o = 16; o > 0; o >>= 1) { s += __shfl_down(s, o); q += __shfl_down(q, o); }
    if (lane == 0) { aux[w] = s; aux[8 + w] = q; }
    __syncthreads();
    if ((tid & 127) == 0) {
        int nd = tid >> 7;
        float S = 0.0f, Q = 0.0f;
        for (int i = 0; i < 4; ++i) { S += aux[nd * 4 + i]; Q += aux[8 + nd * 4 + i]; }
        float mean = S * (1.0f / 128.0f);
        aux[16 + 2 * nd] = mean;
        aux[17 + 2 * nd] = rsqrtf(Q * (1.0f / 128.0f) - mean * mean + EPSF);
    }
    __syncthreads();
    {
        int nd = tid >> 7, f = tid & 127;
        float y = (red[tid] - aux[16 + 2 * nd]) * aux[17 + 2 * nd] * ln1w[f] + ln1b[f];
        hv[(size_t)(n0 + nd) * CF + f] = (_Float16)y;
    }
}

// ---------------------------------------------------------------------------
// Kernel B: position-wise dense MLP 128->512->128 (exact gelu), residual,
// LayerNorm2, output mask. 64-node tile per workgroup (8 wave32).
// h_v tile is staged with CDNA5 async global->LDS copies (ASYNCcnt path).
// ---------------------------------------------------------------------------
__global__ __launch_bounds__(256) void dense_kernel(
    const _Float16* __restrict__ hv,
    const _Float16* __restrict__ pWd1, const float* __restrict__ bd1,
    const _Float16* __restrict__ pWd2, const float* __restrict__ bd2,
    const float* __restrict__ ln2w, const float* __restrict__ ln2b,
    const float* __restrict__ maskp,
    float* __restrict__ out)
{
    extern __shared__ char smem_raw[];
    const int ASTR = 136;   // 64 x 136 halves : h_v tile
    const int HSTR = 520;   // 64 x 520 halves : hidden (4C)
    const int PSTR = 132;   // 64 x 132 floats : pre-LN2
    _Float16* Ah  = (_Float16*)smem_raw;
    _Float16* H1  = Ah + 64 * ASTR;
    float*    P   = (float*)(H1 + 64 * HSTR);
    float*    lnS = P + 64 * PSTR;      // 64 x 4
    float*    lnQ = lnS + 256;          // 64 x 4
    float*    lnM = lnQ + 256;          // 64
    float*    lnI = lnM + 64;           // 64

    const int nb0  = blockIdx.x * 64;
    const int tid  = threadIdx.x;
    const int w    = tid >> 5;
    const int lane = tid & 31;
    const int colL = lane & 15;
    const int rsel = (lane < 16) ? 0 : 8;

    // ---- stage h_v tile via async global->LDS (16B per lane per pass) -----
    // 64 rows x 256B; generic __shared__ pointers keep the LDS byte offset in
    // their low 32 bits, which is what the async op's LDS-address VGPR wants.
    #pragma unroll
    for (int pass = 0; pass < 4; ++pass) {
        int u  = pass * 256 + tid;          // 16B chunk index
        int r  = u >> 4;
        int c0 = (u & 15) * 16;
        unsigned lds_addr = (unsigned)(size_t)&Ah[r * ASTR + c0];
        const _Float16* gptr = hv + (size_t)(nb0 + r) * CF + c0;
        asm volatile("global_load_async_to_lds_b128 %0, %1, off"
                     :: "v"(lds_addr), "v"(gptr)
                     : "memory");
    }
    asm volatile("s_wait_asynccnt 0" ::: "memory");
    __syncthreads();

    // ---- GEMM d1: [64 x 128] @ [128 x 512], gelu -> H1 --------------------
    for (int j = 0; j < 4; ++j) {
        int strip = j * 8 + w;          // 32 column strips of 16
        v8f acc[4] = {};
        #pragma unroll
        for (int kb = 0; kb < 4; ++kb) {
            v16h b = *(const v16h*)(pWd1 + (((kb << 5) + strip) << 9) + lane * 16);
            #pragma unroll
            for (int mt = 0; mt < 4; ++mt) {
                AFrag a;
                int row  = mt * 16 + colL;
                int off0 = row * ASTR + kb * 32 + rsel;
                a.h[0] = *(const v8h*)(Ah + off0);
                a.h[1] = *(const v8h*)(Ah + off0 + 16);
                acc[mt] = __builtin_amdgcn_wmma_f32_16x16x32_f16(
                    false, a.v, false, b, (short)0, acc[mt], false, false);
            }
        }
        int cc = strip * 16 + colL;
        float bias = bd1[cc];
        #pragma unroll
        for (int mt = 0; mt < 4; ++mt)
            #pragma unroll
            for (int i = 0; i < 8; ++i) {
                int row = mt * 16 + i + rsel;
                H1[row * HSTR + cc] = (_Float16)gelu_exact(acc[mt][i] + bias);
            }
    }
    __syncthreads();

    // ---- GEMM d2: [64 x 512] @ [512 x 128] + residual ---------------------
    {
        v8f acc[4] = {};
        #pragma unroll
        for (int kb = 0; kb < 16; ++kb) {
            v16h b = *(const v16h*)(pWd2 + (((kb << 3) + w) << 9) + lane * 16);
            #pragma unroll
            for (int mt = 0; mt < 4; ++mt) {
                AFrag a;
                int row  = mt * 16 + colL;
                int off0 = row * HSTR + kb * 32 + rsel;
                a.h[0] = *(const v8h*)(H1 + off0);
                a.h[1] = *(const v8h*)(H1 + off0 + 16);
                acc[mt] = __builtin_amdgcn_wmma_f32_16x16x32_f16(
                    false, a.v, false, b, (short)0, acc[mt], false, false);
            }
        }
        int cc = w * 16 + colL;
        float bias = bd2[cc];
        #pragma unroll
        for (int mt = 0; mt < 4; ++mt)
            #pragma unroll
            for (int i = 0; i < 8; ++i) {
                int row = mt * 16 + i + rsel;
                P[row * PSTR + cc] = acc[mt][i] + bias + (float)Ah[row * ASTR + cc];
            }
    }
    __syncthreads();

    // ---- LayerNorm2 + output mask -----------------------------------------
    int row = tid >> 2, sub = tid & 3;
    float s = 0.0f, q = 0.0f;
    for (int c = sub * 32; c < sub * 32 + 32; ++c) {
        float v = P[row * PSTR + c];
        s += v; q += v * v;
    }
    lnS[row * 4 + sub] = s; lnQ[row * 4 + sub] = q;
    __syncthreads();
    if (sub == 0) {
        float S = lnS[row*4] + lnS[row*4+1] + lnS[row*4+2] + lnS[row*4+3];
        float Q = lnQ[row*4] + lnQ[row*4+1] + lnQ[row*4+2] + lnQ[row*4+3];
        float mean = S * (1.0f / 128.0f);
        lnM[row] = mean;
        lnI[row] = rsqrtf(Q * (1.0f / 128.0f) - mean * mean + EPSF);
    }
    __syncthreads();
    {
        float mean = lnM[row], inv = lnI[row];
        float mk = maskp[nb0 + row];
        for (int c = sub * 32; c < sub * 32 + 32; ++c) {
            float y = (P[row * PSTR + c] - mean) * inv * ln2w[c] + ln2b[c];
            __builtin_nontemporal_store(mk * y, &out[(size_t)(nb0 + row) * CF + c]);
        }
    }
}

// ---------------------------------------------------------------------------
extern "C" void kernel_launch(void* const* d_in, const int* in_sizes, int n_in,
                              void* d_out, int out_size, void* d_ws, size_t ws_size,
                              hipStream_t stream) {
    const float* nodef = (const float*)d_in[0];
    const float* edgef = (const float*)d_in[1];
    const float* maskp = (const float*)d_in[2];
    const float* attn  = (const float*)d_in[3];
    const float* W1  = (const float*)d_in[4];
    const float* b1  = (const float*)d_in[5];
    const float* W2  = (const float*)d_in[6];
    const float* b2  = (const float*)d_in[7];
    const float* W3  = (const float*)d_in[8];
    const float* b3  = (const float*)d_in[9];
    const float* l1w = (const float*)d_in[10];
    const float* l1b = (const float*)d_in[11];
    const float* Wd1 = (const float*)d_in[12];
    const float* bd1 = (const float*)d_in[13];
    const float* Wd2 = (const float*)d_in[14];
    const float* bd2 = (const float*)d_in[15];
    const float* l2w = (const float*)d_in[16];
    const float* l2b = (const float*)d_in[17];
    float* out = (float*)d_out;

    // workspace layout (halves, 2B each)
    char* ws = (char*)d_ws;
    _Float16* pW1  = (_Float16*)(ws);                 // 512*128
    _Float16* pW2  = (_Float16*)(ws + 131072);        // 128*128
    _Float16* pW3  = (_Float16*)(ws + 163840);        // 128*128
    _Float16* pWd1 = (_Float16*)(ws + 196608);        // 128*512
    _Float16* pWd2 = (_Float16*)(ws + 327680);        // 512*128
    _Float16* hv   = (_Float16*)(ws + 458752);        // 8192*128 f16

    pack_weights<<<(512*128 + 255) / 256, 256, 0, stream>>>(W1,  pW1,  512, 128);
    pack_weights<<<(128*128 + 255) / 256, 256, 0, stream>>>(W2,  pW2,  128, 128);
    pack_weights<<<(128*128 + 255) / 256, 256, 0, stream>>>(W3,  pW3,  128, 128);
    pack_weights<<<(128*512 + 255) / 256, 256, 0, stream>>>(Wd1, pWd1, 128, 512);
    pack_weights<<<(512*128 + 255) / 256, 256, 0, stream>>>(Wd2, pWd2, 512, 128);

    size_t shA = (size_t)(96*520 + 2*96*136) * 2 + (size_t)(256 + 20 + 96) * 4;
    size_t shB = (size_t)(64*136 + 64*520) * 2 + (size_t)(64*132 + 256 + 256 + 64 + 64) * 4;

    msg_kernel<<<N_NODES / 2, 256, shA, stream>>>(nodef, edgef, attn,
                                                  pW1, b1, pW2, b2, pW3, b3,
                                                  l1w, l1b, hv);
    dense_kernel<<<N_NODES / 64, 256, shB, stream>>>(hv, pWd1, bd1, pWd2, bd2,
                                                     l2w, l2b, maskp, out);
}
